// NearestNeighborGraph_21646635172272
// MI455X (gfx1250) — matrix operand order repaired and checked by reading, not compiled, not run on hardware
//
#include <hip/hip_runtime.h>

// ---------------- problem constants (from reference) ----------------
#define NB     8         // batches
#define NPTS   4096      // points per batch
#define DIM    128       // feature dim
#define ROWB   (2*DIM)   // interleaved [hi(128) | lo(128)] bf16 per row
#define KNN    16        // top-k
#define NWAVES 4         // waves per workgroup
#define WGSIZE (NWAVES * 32)
#define NTILES (NPTS / 16)        // 256 column tiles
#define TPW    (NTILES / NWAVES)  // 64 tiles per wave

typedef __attribute__((ext_vector_type(16))) __bf16 v16bf;
typedef __attribute__((ext_vector_type(8)))  __bf16 v8bf;
typedef __attribute__((ext_vector_type(4)))  __bf16 v4bf;
typedef __attribute__((ext_vector_type(8)))  float  v8f;
typedef unsigned int u32;
typedef __attribute__((ext_vector_type(4))) u32 u32x4;
typedef __attribute__((ext_vector_type(8))) int i32x8;
typedef __attribute__((ext_vector_type(4))) int i32x4;

// One-time pre-converted data: bf16 hi/lo split of h + row squared norms.
__device__ __attribute__((aligned(16))) __bf16 g_hs[(size_t)NB * NPTS * ROWB];
__device__ float g_x2[NB * NPTS];

// ---- lane compare-exchange on (value,index) pairs, xor-partner network ----
__device__ __forceinline__ void cmpx(float &v, int &ix, int m, bool keepMin) {
    float ov = __shfl_xor(v, m, 32);
    int   oi = __shfl_xor(ix, m, 32);
    bool take = keepMin ? (ov < v) : (ov > v);
    v  = take ? ov : v;
    ix = take ? oi : ix;
}

// Bitonic sort 16 values ascending across each 16-lane half (10 stages).
__device__ __forceinline__ void sort16(float &v, int &ix, int lane15) {
#pragma unroll
    for (int k = 2; k <= 8; k <<= 1)
#pragma unroll
        for (int j = k >> 1; j >= 1; j >>= 1)
            cmpx(v, ix, j, ((lane15 & k) == 0) == ((lane15 & j) == 0));
#pragma unroll
    for (int j = 8; j >= 1; j >>= 1)
        cmpx(v, ix, j, (lane15 & j) == 0);
}

// Merge sorted-asc state S with sorted-asc T: keep 16 smallest, sorted-asc.
__device__ __forceinline__ void merge16(float &Sv, int &Si, float Tv, int Ti,
                                        int lane15) {
    float rv = __shfl_xor(Tv, 15, 32);
    int   ri = __shfl_xor(Ti, 15, 32);
    if (rv < Sv) { Sv = rv; Si = ri; }
#pragma unroll
    for (int j = 8; j >= 1; j >>= 1)
        cmpx(Sv, Si, j, (lane15 & j) == 0);
}

// Build a v16bf WMMA operand from two 8-element (16 B) chunks.
__device__ __forceinline__ v16bf ldpair(const __bf16 *p0, const __bf16 *p1) {
    v8bf a0 = *(const v8bf *)p0;
    v8bf a1 = *(const v8bf *)p1;
    return __builtin_shufflevector(a0, a1, 0, 1, 2, 3, 4, 5, 6, 7,
                                   8, 9, 10, 11, 12, 13, 14, 15);
}

// ---- Tensor Data Mover: load one 16-row x 512-byte tile into LDS ----
// D# per CDNA5 ISA ch.8: group0 = {flags/count, lds_addr, gaddr lo, gaddr hi
// + type=2}; group1 = {data_size=2B, tensor dims/strides, tile 256x16}.
__device__ __forceinline__ void tdm_load_tile(u32 lds_off, const __bf16 *gsrc) {
    unsigned long long ga = (unsigned long long)(const void *)gsrc;
    u32x4 g0;
    g0.x = 1u;                                     // count=1, user descriptor
    g0.y = lds_off;                                // LDS byte address
    g0.z = (u32)ga;                                // global addr [31:0]
    g0.w = ((u32)(ga >> 32) & 0x01FFFFFFu) | (2u << 30);  // [56:32], type=2
    i32x8 g1;
    g1[0] = (1 << 16);                             // data_size=1 -> 2 bytes
    g1[1] = (ROWB & 0xFFFF) << 16;                 // tensor_dim0 = 256
    g1[2] = ((NPTS & 0xFFFF) << 16);               // tensor_dim1 = 4096
    g1[3] = (ROWB << 16);                          // tile_dim0 = 256
    g1[4] = 16;                                    // tile_dim1 = 16 rows
    g1[5] = ROWB;                                  // tensor_dim0_stride = 256
    g1[6] = 0;
    g1[7] = 0;
    i32x4 z4 = {0, 0, 0, 0};
#if __clang_major__ >= 23
    i32x8 z8 = {0, 0, 0, 0, 0, 0, 0, 0};
    __builtin_amdgcn_tensor_load_to_lds(g0, g1, z4, z4, z8, 0);
#else
    __builtin_amdgcn_tensor_load_to_lds(g0, g1, z4, z4, 0);
#endif
}

// ---- prep: one-time f32 -> bf16 hi/lo split + row norms ----
__global__ __launch_bounds__(256)
void knn_prep(const float *__restrict__ h) {
    const int row  = blockIdx.x * 8 + (threadIdx.x >> 5);
    const int lane = threadIdx.x & 31;
    const float4 v = ((const float4 *)(h + (size_t)row * DIM))[lane];
    float s = v.x * v.x + v.y * v.y + v.z * v.z + v.w * v.w;
    __bf16 hx = (__bf16)v.x, hy = (__bf16)v.y;
    __bf16 hz = (__bf16)v.z, hw = (__bf16)v.w;
    v4bf hv = {hx, hy, hz, hw};
    v4bf lv = {(__bf16)(v.x - (float)hx), (__bf16)(v.y - (float)hy),
               (__bf16)(v.z - (float)hz), (__bf16)(v.w - (float)hw)};
    *(v4bf *)&g_hs[(size_t)row * ROWB + 4 * lane] = hv;
    *(v4bf *)&g_hs[(size_t)row * ROWB + DIM + 4 * lane] = lv;
#pragma unroll
    for (int m = 16; m >= 1; m >>= 1) s += __shfl_xor(s, m, 32);
    if (lane == 0) g_x2[row] = s;
}

// dynamic LDS layout (no static __shared__ -> offsets start at 0)
#define B_TILE_BYTES  (16 * ROWB * 2)                    // 8 KB per tile
#define B_BYTES       (NWAVES * 2 * B_TILE_BYTES)        // 64 KB
#define MV_BYTES      (NWAVES * 16 * KNN * 4)            // 4 KB
#define SMEM_BYTES    (B_BYTES + 2 * MV_BYTES)           // 72 KB

__global__ __launch_bounds__(WGSIZE)
void knn_main(float *__restrict__ out) {
    extern __shared__ __align__(16) char smem[];
    __bf16 *Bb = (__bf16 *)smem;                          // [NWAVES][2][16][ROWB]
    float (*mval)[16][KNN] = (float (*)[16][KNN])(smem + B_BYTES);
    int   (*midx)[16][KNN] = (int (*)[16][KNN])(smem + B_BYTES + MV_BYTES);

    const int tid    = threadIdx.x;
    const int wave   = tid >> 5;
    const int lane   = tid & 31;
    const int lane15 = lane & 15;
    const int sub    = lane >> 4;
    const int m0     = blockIdx.x * 16;
    const int b      = blockIdx.y;
    const int offA   = sub * 8;
    const int offB   = sub * 16;

    // ---- A operands held in registers for the whole kernel (reused 256x) ----
    const __bf16 *arow = g_hs + (size_t)(b * NPTS + m0 + lane15) * ROWB;
    v16bf Ah[4], Al[4];
#pragma unroll
    for (int c = 0; c < 4; ++c) {
        const int kc = 32 * c;
        Ah[c] = ldpair(arow + kc + offA, arow + kc + 16 + offA);
        Al[c] = ldpair(arow + DIM + kc + offA, arow + DIM + kc + 16 + offA);
    }
    const float *x2b = g_x2 + b * NPTS;
    float x2Ar[8];
#pragma unroll
    for (int r = 0; r < 8; ++r) x2Ar[r] = x2b[m0 + r + 8 * sub];

    float Sv[8];
    int   Si[8];
#pragma unroll
    for (int r = 0; r < 8; ++r) { Sv[r] = 3.4e38f; Si[r] = 0; }

    // ---- double-buffered TDM pipeline over this wave's column tiles ----
    const u32 lds0 = (u32)(wave * 2 * B_TILE_BYTES);
    tdm_load_tile(lds0, g_hs + (size_t)(b * NPTS + wave * 16) * ROWB);
    int buf = 0;
    for (int i = 0; i < TPW; ++i) {
        const int n0 = (wave + i * NWAVES) * 16;
        if (i + 1 < TPW) {   // prefetch next tile into the other buffer
            tdm_load_tile(lds0 + (u32)((buf ^ 1) * B_TILE_BYTES),
                          g_hs + (size_t)(b * NPTS + n0 + NWAVES * 16) * ROWB);
            __builtin_amdgcn_s_wait_tensorcnt((short)1);
        } else {
            __builtin_amdgcn_s_wait_tensorcnt((short)0);
        }
        asm volatile("" ::: "memory");   // keep LDS reads below the wait

        const __bf16 *brow =
            Bb + (size_t)(wave * 2 + buf) * 16 * ROWB + (size_t)lane15 * ROWB;
        const float x2col = x2b[n0 + lane15];

        v8f acc = {0.f, 0.f, 0.f, 0.f, 0.f, 0.f, 0.f, 0.f};
#pragma unroll
        for (int c = 0; c < 4; ++c) {
            const int kc = 32 * c;
            v16bf bh = ldpair(brow + kc + offA, brow + kc + 16 + offA);
            v16bf bl = ldpair(brow + DIM + kc + offA, brow + DIM + kc + 16 + offA);
            acc = __builtin_amdgcn_wmma_f32_16x16x32_bf16(
                false, Ah[c], false, bh, (short)0, acc, false, false);
            acc = __builtin_amdgcn_wmma_f32_16x16x32_bf16(
                false, Ah[c], false, bl, (short)0, acc, false, false);
            acc = __builtin_amdgcn_wmma_f32_16x16x32_bf16(
                false, Al[c], false, bh, (short)0, acc, false, false);
        }

        // ---- distances + threshold-gated top-K merge ----
#pragma unroll
        for (int r = 0; r < 8; ++r) {
            float d = fmaf(-2.0f, acc[r], x2Ar[r] + x2col);
            float smax = __shfl(Sv[r], (lane & 16) | 15, 32);
            if (__ballot(d < smax)) {   // wave-uniform gate
                int di = n0 + lane15;
                sort16(d, di, lane15);
                merge16(Sv[r], Si[r], d, di, lane15);
            }
        }
        buf ^= 1;
    }

    // ---- cross-wave merge through LDS ----
#pragma unroll
    for (int r = 0; r < 8; ++r) {
        const int row = r + 8 * sub;
        mval[wave][row][lane15] = Sv[r];
        midx[wave][row][lane15] = Si[r];
    }
    __syncthreads();

    if (wave == 0) {
#pragma unroll
        for (int r = 0; r < 8; ++r) {
            const int row = r + 8 * sub;
            float v = mval[0][row][lane15];
            int   i = midx[0][row][lane15];
            for (int w = 1; w < NWAVES; ++w)
                merge16(v, i, mval[w][row][lane15], midx[w][row][lane15],
                        lane15);
            const int grow = m0 + row;
            const size_t o = ((size_t)b * NPTS + grow) * KNN + lane15;
            out[o] = v;                                        // knn_dist
            out[(size_t)NB * NPTS * KNN + o] = (float)i;       // k_indices
            if (b == 0)                                        // src ids
                out[(size_t)2 * NB * NPTS * KNN +
                    (size_t)grow * KNN + lane15] = (float)grow;
        }
    }
}

extern "C" void kernel_launch(void *const *d_in, const int *in_sizes, int n_in,
                              void *d_out, int out_size, void *d_ws,
                              size_t ws_size, hipStream_t stream) {
    const float *h = (const float *)d_in[0];   // (8, 4096, 128) f32
    float *out = (float *)d_out;               // dist | indices | src (flat)

    // one-time bf16 hi/lo split + row norms (same stream -> ordered)
    hipLaunchKernelGGL(knn_prep, dim3((NB * NPTS) / 8), dim3(256), 0, stream,
                       h);
    // fused WMMA distance + top-k, TDM double-buffered B staging
    hipLaunchKernelGGL(knn_main, dim3(NTILES, NB), dim3(WGSIZE), SMEM_BYTES,
                       stream, out);
}
